// TermAttentionEnhancedBERT_42185168782010
// MI455X (gfx1250) — matrix operand order repaired and checked by reading, not compiled, not run on hardware
//
#include <hip/hip_runtime.h>
#include <math.h>

#define LNUM   12
#define HDIM   768
#define NHEAD  12
#define HEADD  64
#define FFDIM  3072
#define SEQ    512
#define BATCH  16
#define NTOK   (BATCH*SEQ)      // 8192
#define COMB   (HDIM + HDIM/4)  // 960

typedef __attribute__((ext_vector_type(16))) __bf16 v16b;
typedef __attribute__((ext_vector_type(8)))  __bf16 v8b;
typedef __attribute__((ext_vector_type(4)))  __bf16 v4b;
typedef __attribute__((ext_vector_type(8)))  float  v8f;
typedef __attribute__((ext_vector_type(4)))  float  v4f;

__device__ inline v8f vzero8() {
    v8f z = {0.f,0.f,0.f,0.f,0.f,0.f,0.f,0.f};
    return z;
}

// ---- WMMA fragment loaders (CDNA5 16x16x32 bf16 layouts, wave32) ----
__device__ inline v16b ld_afrag_bfv(const __bf16* t, int ld, int row0, int k0) {
    int lane = threadIdx.x & 31;
    int half = lane >> 4;
    int m = row0 + (lane & 15);
    const __bf16* p = t + (size_t)m*ld + k0 + half*8;
    v8b lo = *(const v8b*)(p);
    v8b hi = *(const v8b*)(p + 16);
    return __builtin_shufflevector(lo, hi, 0,1,2,3,4,5,6,7,8,9,10,11,12,13,14,15);
}
__device__ inline v16b ld_afrag_f32v(const float* t, int ld, int row0, int k0) {
    int lane = threadIdx.x & 31;
    int half = lane >> 4;
    int m = row0 + (lane & 15);
    const float* p = t + (size_t)m*ld + k0 + half*8;
    v4f x0 = *(const v4f*)(p);
    v4f x1 = *(const v4f*)(p + 4);
    v4f x2 = *(const v4f*)(p + 16);
    v4f x3 = *(const v4f*)(p + 20);
    v16b a;
    a[0]=(__bf16)x0[0]; a[1]=(__bf16)x0[1]; a[2]=(__bf16)x0[2]; a[3]=(__bf16)x0[3];
    a[4]=(__bf16)x1[0]; a[5]=(__bf16)x1[1]; a[6]=(__bf16)x1[2]; a[7]=(__bf16)x1[3];
    a[8]=(__bf16)x2[0]; a[9]=(__bf16)x2[1]; a[10]=(__bf16)x2[2]; a[11]=(__bf16)x2[3];
    a[12]=(__bf16)x3[0]; a[13]=(__bf16)x3[1]; a[14]=(__bf16)x3[2]; a[15]=(__bf16)x3[3];
    return a;
}
__device__ inline v16b ld_bfrag_t(const __bf16* t, int ld, int col0, int k0) {
    int lane = threadIdx.x & 31;
    int n = col0 + (lane & 15);
    int kb = k0 + (lane >> 4) * 8;
    const __bf16* p = t + (size_t)n*ld + kb;
    v8b lo = *(const v8b*)(p);
    v8b hi = *(const v8b*)(p + 16);
    return __builtin_shufflevector(lo, hi, 0,1,2,3,4,5,6,7,8,9,10,11,12,13,14,15);
}

__device__ inline v8f wmma_bf16(v16b a, v16b b, v8f c) {
    return __builtin_amdgcn_wmma_f32_16x16x32_bf16(false, a, false, b, (short)0, c, false, false);
}

// ---- block reductions (256 threads) ----
__device__ inline float blk_sum256(float v, float* sd) {
    int t = threadIdx.x;
    sd[t] = v; __syncthreads();
    for (int s2 = 128; s2 > 0; s2 >>= 1) {
        if (t < s2) sd[t] += sd[t + s2];
        __syncthreads();
    }
    float r = sd[0]; __syncthreads();
    return r;
}
__device__ inline float blk_max256(float v, float* sd) {
    int t = threadIdx.x;
    sd[t] = v; __syncthreads();
    for (int s2 = 128; s2 > 0; s2 >>= 1) {
        if (t < s2) sd[t] = fmaxf(sd[t], sd[t + s2]);
        __syncthreads();
    }
    float r = sd[0]; __syncthreads();
    return r;
}

// ============ fp32 -> bf16 convert (flat, 4/thread) ============
__global__ __launch_bounds__(256) void cvt_bf16_kernel(
    const float* __restrict__ src, __bf16* __restrict__ dst, int n4)
{
    int i = blockIdx.x*256 + threadIdx.x;
    if (i < n4) {
        v4f v = *(const v4f*)(src + (size_t)i*4);
        v4b o;
        o[0]=(__bf16)v[0]; o[1]=(__bf16)v[1]; o[2]=(__bf16)v[2]; o[3]=(__bf16)v[3];
        *(v4b*)(dst + (size_t)i*4) = o;
    }
}

// ============ fp32 W[K,N] -> bf16 Wt[N,K] (32x32 tiles; K,N multiples of 32) ============
__global__ __launch_bounds__(256) void convt_kernel(
    const float* __restrict__ W, __bf16* __restrict__ Wt, int K, int N)
{
    __shared__ float t[32][33];
    int kt = blockIdx.y*32, nt = blockIdx.x*32;
    int tx = threadIdx.x & 31, ty = threadIdx.x >> 5;   // 32 x 8
#pragma unroll
    for (int i = 0; i < 4; ++i) {
        int k = kt + ty + i*8;
        t[ty + i*8][tx] = W[(size_t)k*N + nt + tx];
    }
    __syncthreads();
#pragma unroll
    for (int i = 0; i < 4; ++i) {
        int n = nt + ty + i*8;
        Wt[(size_t)n*K + kt + tx] = (__bf16)t[tx][ty + i*8];
    }
}

// ============ embeddings + layernorm (dual fp32 + bf16 output) ============
__global__ __launch_bounds__(256) void embed_kernel(
    const int* __restrict__ ids, const int* __restrict__ tt,
    const float* __restrict__ we, const float* __restrict__ pe,
    const float* __restrict__ te, const float* __restrict__ g,
    const float* __restrict__ bb, float* __restrict__ X, __bf16* __restrict__ Xb)
{
    __shared__ float sd[256];
    int i = blockIdx.x;
    int s = i & (SEQ - 1);
    int id = ids[i], ty = tt[i];
    int tid = threadIdx.x;
    float v[3]; float part = 0.f;
#pragma unroll
    for (int j = 0; j < 3; ++j) {
        int e = tid + j*256;
        v[j] = we[(size_t)id*HDIM + e] + pe[(size_t)s*HDIM + e] + te[(size_t)ty*HDIM + e];
        part += v[j];
    }
    float mean = blk_sum256(part, sd) * (1.f/768.f);
    float vp = 0.f;
#pragma unroll
    for (int j = 0; j < 3; ++j) { float d = v[j] - mean; vp += d*d; }
    float var = blk_sum256(vp, sd) * (1.f/768.f);
    float inv = rsqrtf(var + 1e-12f);
#pragma unroll
    for (int j = 0; j < 3; ++j) {
        int e = tid + j*256;
        float r = (v[j]-mean)*inv*g[e] + bb[e];
        X[(size_t)i*HDIM + e]  = r;
        Xb[(size_t)i*HDIM + e] = (__bf16)r;
    }
}

// ============ residual + layernorm (in-place X, plus bf16 copy) ============
__global__ __launch_bounds__(256) void residual_ln_kernel(
    float* __restrict__ X, const float* __restrict__ Y,
    const float* __restrict__ g, const float* __restrict__ bb,
    __bf16* __restrict__ Xb)
{
    __shared__ float sd[256];
    int i = blockIdx.x;
    int tid = threadIdx.x;
    float v[3]; float part = 0.f;
#pragma unroll
    for (int j = 0; j < 3; ++j) {
        int e = tid + j*256;
        v[j] = X[(size_t)i*HDIM + e] + Y[(size_t)i*HDIM + e];
        part += v[j];
    }
    float mean = blk_sum256(part, sd) * (1.f/768.f);
    float vp = 0.f;
#pragma unroll
    for (int j = 0; j < 3; ++j) { float d = v[j] - mean; vp += d*d; }
    float var = blk_sum256(vp, sd) * (1.f/768.f);
    float inv = rsqrtf(var + 1e-12f);
#pragma unroll
    for (int j = 0; j < 3; ++j) {
        int e = tid + j*256;
        float r = (v[j]-mean)*inv*g[e] + bb[e];
        X[(size_t)i*HDIM + e]  = r;
        Xb[(size_t)i*HDIM + e] = (__bf16)r;
    }
}

// ============ full-tile bf16 GEMM (no guards, double-buffered LDS) ============
// 128 threads = 4 waves; block tile 128x128; wave tile 64x64 (4x4 accums).
// 16 ds_load_b128 per 16 WMMA per K-step (1.0 LDS-load/WMMA).
// C[M,N] = act(A[M,lda] * Bt[N,K]^T + bias); M,N multiples of 128, K of 32.
__global__ __launch_bounds__(128) void gemm_full_kernel(
    const __bf16* __restrict__ A, int lda,
    const __bf16* __restrict__ Bt, const float* __restrict__ bias,
    float* __restrict__ C, __bf16* __restrict__ Cb,
    int M, int N, int K, int act)
{
    __shared__ __attribute__((aligned(16))) __bf16 As[2][128*40];
    __shared__ __attribute__((aligned(16))) __bf16 Bs[2][128*40];
    int tid = threadIdx.x;
    int m0 = blockIdx.y * 128, n0 = blockIdx.x * 128;
    int w = tid >> 5;
    int wm = (w & 1) * 64;
    int wn = (w >> 1) * 64;
    v8f acc[4][4];
#pragma unroll
    for (int mi = 0; mi < 4; ++mi)
#pragma unroll
        for (int ni = 0; ni < 4; ++ni) acc[mi][ni] = vzero8();

    auto stage = [&](int buf, int k0) {
#pragma unroll
        for (int i = 0; i < 4; ++i) {
            int gi = tid + i*128;
            int r = gi >> 2, c8 = (gi & 3)*8;
            *(v8b*)(&As[buf][r*40 + c8]) =
                *(const v8b*)(A + (size_t)(m0 + r)*lda + k0 + c8);
        }
#pragma unroll
        for (int i = 0; i < 4; ++i) {
            int gi = tid + i*128;
            int r = gi >> 2, c8 = (gi & 3)*8;
            *(v8b*)(&Bs[buf][r*40 + c8]) =
                *(const v8b*)(Bt + (size_t)(n0 + r)*K + k0 + c8);
        }
    };

    int nk = K >> 5;
    stage(0, 0);
    __syncthreads();
    for (int kt = 0; kt < nk; ++kt) {
        int cur = kt & 1;
        if (kt + 1 < nk) stage(cur ^ 1, (kt + 1)*32);
        if (kt + 2 < nk) {   // warm caches for the tile after next (global_prefetch_b8)
            __builtin_prefetch(A  + (size_t)(m0 + (tid >> 2)*4)*lda + (kt + 2)*32, 0, 1);
            __builtin_prefetch(Bt + (size_t)(n0 + (tid >> 2)*4)*K   + (kt + 2)*32, 0, 1);
        }
        v16b bfr[4];
#pragma unroll
        for (int ni = 0; ni < 4; ++ni) bfr[ni] = ld_bfrag_t(Bs[cur], 40, wn + ni*16, 0);
#pragma unroll
        for (int mi = 0; mi < 4; ++mi) {
            v16b af = ld_afrag_bfv(As[cur], 40, wm + mi*16, 0);
#pragma unroll
            for (int ni = 0; ni < 4; ++ni)
                acc[mi][ni] = wmma_bf16(af, bfr[ni], acc[mi][ni]);
        }
        __syncthreads();
    }

    int lane = tid & 31;
    int cn = lane & 15, mb = (lane >> 4) * 8;
#pragma unroll
    for (int mi = 0; mi < 4; ++mi)
#pragma unroll
        for (int ni = 0; ni < 4; ++ni) {
            int nn = n0 + wn + ni*16 + cn;
            float bia = bias[nn];
#pragma unroll
            for (int r = 0; r < 8; ++r) {
                int mm = m0 + wm + mi*16 + mb + r;
                float v = acc[mi][ni][r] + bia;
                if (act == 1)      v = 0.5f*v*(1.0f + erff(v*0.70710678118f)); // exact gelu
                else if (act == 2) v = tanhf(v);
                if (Cb) Cb[(size_t)mm*N + nn] = (__bf16)v;
                else    C [(size_t)mm*N + nn] = v;
            }
        }
}

// ============ small guarded GEMM (head GEMMs: M=16) ============
__global__ __launch_bounds__(256) void gemm_small_kernel(
    const __bf16* __restrict__ A, int lda,
    const __bf16* __restrict__ Bt, const float* __restrict__ bias,
    float* __restrict__ C, int M, int N, int K, int act)
{
    __shared__ __attribute__((aligned(16))) __bf16 As[128*40];
    __shared__ __attribute__((aligned(16))) __bf16 Bs[128*40];
    int tid = threadIdx.x;
    int m0 = blockIdx.y * 128, n0 = blockIdx.x * 128;
    int w = tid >> 5;
    int wm = (w & 3) * 32;
    int wn = (w >> 2) * 64;
    v8f acc[2][4];
#pragma unroll
    for (int mi = 0; mi < 2; ++mi)
#pragma unroll
        for (int ni = 0; ni < 4; ++ni) acc[mi][ni] = vzero8();

    for (int k0 = 0; k0 < K; k0 += 32) {
#pragma unroll
        for (int i = 0; i < 16; ++i) {
            int idx = tid + i*256;
            int r = idx >> 5, c = idx & 31;
            int gm = m0 + r;
            As[r*40 + c] = (gm < M) ? A[(size_t)gm*lda + k0 + c] : (__bf16)0.f;
        }
#pragma unroll
        for (int i = 0; i < 16; ++i) {
            int idx = tid + i*256;
            int r = idx >> 5, c = idx & 31;
            int gn = n0 + r;
            Bs[r*40 + c] = (gn < N) ? Bt[(size_t)gn*K + k0 + c] : (__bf16)0.f;
        }
        __syncthreads();
        v16b af[2];
        af[0] = ld_afrag_bfv(As, 40, wm,      0);
        af[1] = ld_afrag_bfv(As, 40, wm + 16, 0);
        v16b bf2[4];
#pragma unroll
        for (int ni = 0; ni < 4; ++ni) bf2[ni] = ld_bfrag_t(Bs, 40, wn + ni*16, 0);
#pragma unroll
        for (int mi = 0; mi < 2; ++mi)
#pragma unroll
            for (int ni = 0; ni < 4; ++ni)
                acc[mi][ni] = wmma_bf16(af[mi], bf2[ni], acc[mi][ni]);
        __syncthreads();
    }

    int lane = tid & 31;
    int cn = lane & 15, mb = (lane >> 4) * 8;
#pragma unroll
    for (int mi = 0; mi < 2; ++mi)
#pragma unroll
        for (int ni = 0; ni < 4; ++ni) {
            int nn = n0 + wn + ni*16 + cn;
            float bia = (nn < N) ? bias[nn] : 0.f;
#pragma unroll
            for (int r = 0; r < 8; ++r) {
                int mm = m0 + wm + mi*16 + mb + r;
                if (mm < M && nn < N) {
                    float v = acc[mi][ni][r] + bia;
                    if (act == 1)      v = 0.5f*v*(1.0f + erff(v*0.70710678118f));
                    else if (act == 2) v = tanhf(v);
                    C[(size_t)mm*N + nn] = v;
                }
            }
        }
}

// ============ fused attention (bf16 in / bf16 out) ============
// grid: (SEQ/32, BATCH*NHEAD); 128 threads = 4 waves
__global__ __launch_bounds__(128) void attention_kernel(
    const __bf16* __restrict__ Q, const __bf16* __restrict__ K,
    const __bf16* __restrict__ V, const int* __restrict__ amask,
    const float* __restrict__ head_bias_l, __bf16* __restrict__ ctx)
{
    __shared__ __attribute__((aligned(16))) __bf16 Qs[32*72];
    __shared__ __attribute__((aligned(16))) __bf16 Ks[512*72];  // phase3: reused as Vt[64*40]
    __shared__ __attribute__((aligned(16))) float  Ss[32*520];
    __shared__ float red[32][4];
    __bf16* Vt = Ks;

    int tid = threadIdx.x;
    int bh = blockIdx.y;
    int b = bh / NHEAD, h = bh % NHEAD;
    int q0 = blockIdx.x * 32;
    const __bf16* Qb = Q + (size_t)b*SEQ*HDIM + h*HEADD;
    const __bf16* Kb = K + (size_t)b*SEQ*HDIM + h*HEADD;
    const __bf16* Vb = V + (size_t)b*SEQ*HDIM + h*HEADD;

    // stage Q tile (32x64): straight bf16 copy
    for (int i = tid; i < 256; i += 128) {
        int r = i >> 3, d8 = (i & 7)*8;
        *(v8b*)(Qs + r*72 + d8) = *(const v8b*)(Qb + (size_t)(q0 + r)*HDIM + d8);
    }
    // stage K (512x64): K row-major IS K^T in [n][k] form
    for (int i = tid; i < 4096; i += 128) {
        int n = i >> 3, d8 = (i & 7)*8;
        *(v8b*)(Ks + n*72 + d8) = *(const v8b*)(Kb + (size_t)n*HDIM + d8);
    }
    __syncthreads();

    int w = tid >> 5;
    int lane = tid & 31;
    int wrow = (w & 1) * 16;
    int wcol = (w >> 1) * 256;
    float hb = 0.4f * head_bias_l[h];          // BIAS_SCALE * head_bias

    // phase 1: scores 32x512 = Q * K^T; wave does 16 rows x 256 cols
    v16b aq0 = ld_afrag_bfv(Qs, 72, wrow, 0);
    v16b aq1 = ld_afrag_bfv(Qs, 72, wrow, 32);
    for (int nt = 0; nt < 16; ++nt) {
        int col0 = wcol + nt*16;
        v8f acc = vzero8();
        v16b b0 = ld_bfrag_t(Ks, 72, col0, 0);
        v16b b1 = ld_bfrag_t(Ks, 72, col0, 32);
        acc = wmma_bf16(aq0, b0, acc);
        acc = wmma_bf16(aq1, b1, acc);
        int n = col0 + (lane & 15);
        int mb = (lane >> 4) * 8;
        float mask = (1.0f - (float)amask[b*SEQ + n]) * -10000.0f;
#pragma unroll
        for (int r = 0; r < 8; ++r)
            Ss[(wrow + mb + r)*520 + n] = acc[r]*0.125f + hb + mask;
    }
    __syncthreads();

    // phase 2: softmax per row (4 threads/row), vectorized LDS
    {
        int row = tid >> 2, seg = tid & 3;
        float* rp = Ss + row*520;
        float mx = -1e30f;
        for (int c = seg*128; c < seg*128 + 128; c += 4) {
            v4f v = *(const v4f*)(rp + c);
            mx = fmaxf(mx, fmaxf(fmaxf(v[0], v[1]), fmaxf(v[2], v[3])));
        }
        red[row][seg] = mx; __syncthreads();
        mx = fmaxf(fmaxf(red[row][0], red[row][1]), fmaxf(red[row][2], red[row][3]));
        __syncthreads();
        float sm = 0.f;
        for (int c = seg*128; c < seg*128 + 128; c += 4) {
            v4f v = *(const v4f*)(rp + c);
            v[0] = __expf(v[0]-mx); v[1] = __expf(v[1]-mx);
            v[2] = __expf(v[2]-mx); v[3] = __expf(v[3]-mx);
            *(v4f*)(rp + c) = v;
            sm += v[0]+v[1]+v[2]+v[3];
        }
        red[row][seg] = sm; __syncthreads();
        sm = red[row][0] + red[row][1] + red[row][2] + red[row][3];
        float inv = 1.0f/sm;
        for (int c = seg*128; c < seg*128 + 128; c += 4) {
            v4f v = *(const v4f*)(rp + c);
            v[0]*=inv; v[1]*=inv; v[2]*=inv; v[3]*=inv;
            *(v4f*)(rp + c) = v;
        }
    }
    __syncthreads();

    // phase 3: ctx(32x64) = P(32x512) * V(512x64); wave does 16 rows x 32 cols
    int ncol0 = (w >> 1) * 32;
    v8f c0 = vzero8(), c1 = vzero8();
    for (int ks = 0; ks < 16; ++ks) {
        // stage V tile transposed: Vt[d][r] = V[ks*32+r][d]  (64 x 32, K-contiguous)
        for (int i = tid; i < 512; i += 128) {
            int d = i >> 3, r0 = (i & 7)*4;
            v4b o;
#pragma unroll
            for (int j = 0; j < 4; ++j)
                o[j] = Vb[(size_t)(ks*32 + r0 + j)*HDIM + d];
            *(v4b*)(Vt + d*40 + r0) = o;
        }
        __syncthreads();
        v16b ap = ld_afrag_f32v(Ss, 520, wrow, ks*32);
        v16b b0 = ld_bfrag_t(Vt, 40, ncol0,      0);
        v16b b1 = ld_bfrag_t(Vt, 40, ncol0 + 16, 0);
        c0 = wmma_bf16(ap, b0, c0);
        c1 = wmma_bf16(ap, b1, c1);
        __syncthreads();
    }
    {
        int n0 = ncol0 + (lane & 15);
        int mb = (lane >> 4) * 8;
        __bf16* Ob = ctx + (size_t)b*SEQ*HDIM + h*HEADD;
#pragma unroll
        for (int r = 0; r < 8; ++r) {
            int m = q0 + wrow + mb + r;
            Ob[(size_t)m*HDIM + n0]      = (__bf16)c0[r];
            Ob[(size_t)m*HDIM + n0 + 16] = (__bf16)c1[r];
        }
    }
}

// ============ term-attention head ============
__global__ __launch_bounds__(256) void term_bias_kernel(
    const float* __restrict__ X, const float* __restrict__ Wtb,
    const float* __restrict__ btb, const int* __restrict__ tmask,
    const int* __restrict__ amask, float* __restrict__ out)
{
    __shared__ float sd[256];
    int i = blockIdx.x;
    int tid = threadIdx.x;
    float part = 0.f;
#pragma unroll
    for (int j = 0; j < 3; ++j) {
        int e = tid + j*256;
        part += X[(size_t)i*HDIM + e] * Wtb[e];
    }
    float tot = blk_sum256(part, sd);
    if (tid == 0) {
        float raw = tot + btb[0];
        float scaled = tanhf(raw) * 2.0f;      // TERM_SCALE
        float tb = tmask[i] ? scaled : -5.0f;
        float base = (1.0f - (float)amask[i]) * -10000.0f;
        out[i] = base + tb;
    }
}

__global__ __launch_bounds__(256) void term_softmax_kernel(
    const float* __restrict__ logits, float* __restrict__ wts)
{
    __shared__ float sd[256];
    int b = blockIdx.x, tid = threadIdx.x;
    float a0 = logits[b*SEQ + tid], a1 = logits[b*SEQ + tid + 256];
    float mx = blk_max256(fmaxf(a0, a1), sd);
    float e0 = __expf(a0 - mx), e1 = __expf(a1 - mx);
    float sm = blk_sum256(e0 + e1, sd);
    float inv = 1.0f/sm;
    wts[b*SEQ + tid]       = e0*inv;
    wts[b*SEQ + tid + 256] = e1*inv;
}

__global__ __launch_bounds__(256) void term_pool_kernel(
    const float* __restrict__ X, const float* __restrict__ wts,
    float* __restrict__ TP)
{
    int b = blockIdx.x;
    int e = blockIdx.y*256 + threadIdx.x;
    float acc = 0.f;
    for (int s = 0; s < SEQ; ++s)
        acc += X[((size_t)b*SEQ + s)*HDIM + e] * wts[b*SEQ + s];
    TP[(size_t)b*HDIM + e] = acc;
}

__global__ __launch_bounds__(256) void head_out_kernel(
    const float* __restrict__ pooled, const float* __restrict__ tf,
    const float* __restrict__ Wc, const float* __restrict__ bc,
    float* __restrict__ out)
{
    __shared__ float comb[COMB];
    int b = blockIdx.x, tid = threadIdx.x;
    for (int i = tid; i < HDIM; i += 256)    comb[i]        = pooled[(size_t)b*HDIM + i];
    for (int i = tid; i < HDIM/4; i += 256)  comb[HDIM + i] = tf[(size_t)b*(HDIM/4) + i];
    __syncthreads();
    for (int i = tid; i < COMB; i += 256)
        out[32 + (size_t)b*COMB + i] = comb[i];           // combined (2nd output)
    if (tid < 2) {
        float s = bc[tid];
        for (int c = 0; c < COMB; ++c) s += comb[c]*Wc[c*2 + tid];
        out[b*2 + tid] = s;                                // logits (1st output)
    }
}

// ============ host side ============
static void launch_gemm_full(const __bf16* A, int lda, const __bf16* Bt, const float* bias,
                             float* C, __bf16* Cb, int M, int N, int K, int act, hipStream_t s)
{
    dim3 g(N/128, M/128), blk(128);
    gemm_full_kernel<<<g, blk, 0, s>>>(A, lda, Bt, bias, C, Cb, M, N, K, act);
}
static void launch_gemm_small(const __bf16* A, int lda, const __bf16* Bt, const float* bias,
                              float* C, int M, int N, int K, int act, hipStream_t s)
{
    dim3 g((N + 127)/128, (M + 127)/128), blk(256);
    gemm_small_kernel<<<g, blk, 0, s>>>(A, lda, Bt, bias, C, M, N, K, act);
}
static void launch_convt(const float* W, __bf16* Wt, int K, int N, hipStream_t s)
{
    convt_kernel<<<dim3(N/32, K/32), 256, 0, s>>>(W, Wt, K, N);
}

extern "C" void kernel_launch(void* const* d_in, const int* in_sizes, int n_in,
                              void* d_out, int out_size, void* d_ws, size_t ws_size,
                              hipStream_t stream)
{
    (void)in_sizes; (void)n_in; (void)out_size; (void)ws_size;
    const int*   input_ids = (const int*)  d_in[0];
    const int*   amask     = (const int*)  d_in[1];
    const int*   ttids     = (const int*)  d_in[2];
    const int*   tmask     = (const int*)  d_in[3];
    const float* word_emb  = (const float*)d_in[4];
    const float* pos_emb   = (const float*)d_in[5];
    const float* type_emb  = (const float*)d_in[6];
    const float* emb_ln_g  = (const float*)d_in[7];
    const float* emb_ln_b  = (const float*)d_in[8];
    const float* Wq  = (const float*)d_in[9];
    const float* bq  = (const float*)d_in[10];
    const float* Wk  = (const float*)d_in[11];
    const float* bk  = (const float*)d_in[12];
    const float* Wv  = (const float*)d_in[13];
    const float* bv  = (const float*)d_in[14];
    const float* hbias = (const float*)d_in[15];
    const float* Wo  = (const float*)d_in[16];
    const float* bo  = (const float*)d_in[17];
    const float* ln1g = (const float*)d_in[18];
    const float* ln1b = (const float*)d_in[19];
    const float* Wff1 = (const float*)d_in[20];
    const float* bff1 = (const float*)d_in[21];
    const float* Wff2 = (const float*)d_in[22];
    const float* bff2 = (const float*)d_in[23];
    const float* ln2g = (const float*)d_in[24];
    const float* ln2b = (const float*)d_in[25];
    const float* Wp  = (const float*)d_in[26];
    const float* bp  = (const float*)d_in[27];
    const float* Wtb = (const float*)d_in[28];
    const float* btb = (const float*)d_in[29];
    const float* Wtp = (const float*)d_in[30];
    const float* btp = (const float*)d_in[31];
    const float* Wc  = (const float*)d_in[32];
    const float* bc  = (const float*)d_in[33];
    float* out = (float*)d_out;

    // workspace layout (256B-aligned chunks)
    uint8_t* base = (uint8_t*)d_ws;
    size_t off = 0;
    auto alloc = [&](size_t bytes) -> void* {
        void* p = base + off;
        off = (off + bytes + 255) & ~(size_t)255;
        return p;
    };
    const size_t NH32 = (size_t)NTOK * HDIM;   // 6,291,456 elems
    float*  X    = (float*) alloc(NH32*4);
    float*  Qf   = (float*) alloc(NH32*4);                    // fp32 GEMM out (pre-LN)
    __bf16* Xb   = (__bf16*)alloc(NH32*2);
    __bf16* Qbb  = (__bf16*)alloc(NH32*2);
    __bf16* Kbb  = (__bf16*)alloc(NH32*2);
    __bf16* Vbb  = (__bf16*)alloc(NH32*2);
    __bf16* CTXb = (__bf16*)alloc(NH32*2);
    __bf16* T1b  = (__bf16*)alloc((size_t)NTOK*FFDIM*2);      // ffn hidden (bf16)
    __bf16* BT   = (__bf16*)alloc((size_t)HDIM*FFDIM*2);      // transposed weight scratch
    float*  RAW  = (float*) alloc(NTOK*4);
    float*  WTS  = (float*) alloc(NTOK*4);
    float*  POOL = (float*) alloc((size_t)BATCH*HDIM*4);
    float*  TP   = (float*) alloc((size_t)BATCH*HDIM*4);
    __bf16* TPb  = (__bf16*)alloc((size_t)BATCH*HDIM*2);
    float*  TF   = (float*) alloc((size_t)BATCH*(HDIM/4)*4);

    // embeddings + LN (writes X fp32 + Xb bf16)
    embed_kernel<<<NTOK, 256, 0, stream>>>(input_ids, ttids, word_emb, pos_emb,
                                           type_emb, emb_ln_g, emb_ln_b, X, Xb);

    for (int l = 0; l < LNUM; ++l) {
        const float* Wq_l = Wq + (size_t)l*HDIM*HDIM;
        const float* Wk_l = Wk + (size_t)l*HDIM*HDIM;
        const float* Wv_l = Wv + (size_t)l*HDIM*HDIM;
        const float* Wo_l = Wo + (size_t)l*HDIM*HDIM;
        const float* W1_l = Wff1 + (size_t)l*HDIM*FFDIM;
        const float* W2_l = Wff2 + (size_t)l*FFDIM*HDIM;

        launch_convt(Wq_l, BT, HDIM, HDIM, stream);
        launch_gemm_full(Xb, HDIM, BT, bq + l*HDIM, nullptr, Qbb, NTOK, HDIM, HDIM, 0, stream);
        launch_convt(Wk_l, BT, HDIM, HDIM, stream);
        launch_gemm_full(Xb, HDIM, BT, bk + l*HDIM, nullptr, Kbb, NTOK, HDIM, HDIM, 0, stream);
        launch_convt(Wv_l, BT, HDIM, HDIM, stream);
        launch_gemm_full(Xb, HDIM, BT, bv + l*HDIM, nullptr, Vbb, NTOK, HDIM, HDIM, 0, stream);

        attention_kernel<<<dim3(SEQ/32, BATCH*NHEAD), 128, 0, stream>>>(
            Qbb, Kbb, Vbb, amask, hbias + l*NHEAD, CTXb);

        launch_convt(Wo_l, BT, HDIM, HDIM, stream);
        launch_gemm_full(CTXb, HDIM, BT, bo + l*HDIM, Qf, nullptr, NTOK, HDIM, HDIM, 0, stream);
        residual_ln_kernel<<<NTOK, 256, 0, stream>>>(X, Qf, ln1g + l*HDIM, ln1b + l*HDIM, Xb);

        launch_convt(W1_l, BT, HDIM, FFDIM, stream);
        launch_gemm_full(Xb, HDIM, BT, bff1 + l*FFDIM, nullptr, T1b, NTOK, FFDIM, HDIM, 1, stream);
        launch_convt(W2_l, BT, FFDIM, HDIM, stream);
        launch_gemm_full(T1b, FFDIM, BT, bff2 + l*HDIM, Qf, nullptr, NTOK, HDIM, FFDIM, 0, stream);
        residual_ln_kernel<<<NTOK, 256, 0, stream>>>(X, Qf, ln2g + l*HDIM, ln2b + l*HDIM, Xb);
    }

    // pooler: tanh(x[:,0] @ Wp + bp) — CLS rows strided by SEQ*HDIM in bf16 copy
    launch_convt(Wp, BT, HDIM, HDIM, stream);
    launch_gemm_small(Xb, SEQ*HDIM, BT, bp, POOL, BATCH, HDIM, HDIM, 2, stream);

    // term attention pooling
    term_bias_kernel<<<NTOK, 256, 0, stream>>>(X, Wtb, btb, tmask, amask, RAW);
    term_softmax_kernel<<<BATCH, 256, 0, stream>>>(RAW, WTS);
    term_pool_kernel<<<dim3(BATCH, HDIM/256), 256, 0, stream>>>(X, WTS, TP);
    {
        int n4 = (int)(((size_t)BATCH*HDIM)/4);
        cvt_bf16_kernel<<<(n4 + 255)/256, 256, 0, stream>>>(TP, TPb, n4);
    }
    launch_convt(Wtp, BT, HDIM, HDIM/4, stream);
    launch_gemm_small(TPb, HDIM, BT, btp, TF, BATCH, HDIM/4, HDIM, 0, stream);

    // concat + classifier -> (logits, combined)
    head_out_kernel<<<BATCH, 256, 0, stream>>>(POOL, TF, Wc, bc, out);
}
